// InvariantValueNetwork_71184787964018
// MI455X (gfx1250) — matrix operand order repaired and checked by reading, not compiled, not run on hardware
//
#include <hip/hip_runtime.h>
#include <hip/hip_bf16.h>

#define NN 100000
#define NE 1600000

typedef __attribute__((ext_vector_type(8)))  float          v8f;
typedef __attribute__((ext_vector_type(16))) __bf16         v16bf;
typedef __attribute__((ext_vector_type(8)))  unsigned short v8u16;
typedef __attribute__((ext_vector_type(16))) unsigned short v16u16;
typedef __attribute__((ext_vector_type(8)))  unsigned int   v8u32;

// f32x2 -> packed bf16x2 in one v_perm_b32 (round-half-up via +0x8000)
__device__ __forceinline__ unsigned pk2(float a, float b) {
  unsigned ua = __builtin_bit_cast(unsigned, a) + 0x8000u;
  unsigned ub = __builtin_bit_cast(unsigned, b) + 0x8000u;
  // dst bytes {3,2} = ub bytes {3,2};  dst bytes {1,0} = ua bytes {3,2}
  return __builtin_amdgcn_perm(ub, ua, 0x07060302u);
}

__device__ __forceinline__ unsigned short bf1(float f) {
  return (unsigned short)((__builtin_bit_cast(unsigned, f) + 0x8000u) >> 16);
}

__device__ __forceinline__ v16bf joinB(v8u16 lo, v8u16 hi) {
  return __builtin_bit_cast(v16bf, __builtin_shufflevector(
      lo, hi, 0,1,2,3,4,5,6,7,8,9,10,11,12,13,14,15));
}

// B-fragment: bf16 along K at p[0..7] and p[16..23] for this lane's column (two ds_load_b128)
__device__ __forceinline__ v16bf ldsB(const unsigned short* p) {
  v8u16 lo = *(const v8u16*)p;
  v8u16 hi = *(const v8u16*)(p + 16);
  return joinB(lo, hi);
}

__device__ __forceinline__ v16bf packA(float4 a, float4 b, float4 c, float4 d) {
  v8u32 r;
  r[0] = pk2(a.x, a.y); r[1] = pk2(a.z, a.w);
  r[2] = pk2(b.x, b.y); r[3] = pk2(b.z, b.w);
  r[4] = pk2(c.x, c.y); r[5] = pk2(c.z, c.w);
  r[6] = pk2(d.x, d.y); r[7] = pk2(d.z, d.w);
  return __builtin_bit_cast(v16bf, r);
}

__device__ __forceinline__ v8f wmma_bf16(v16bf a, v16bf b, v8f c) {
  return __builtin_amdgcn_wmma_f32_16x16x32_bf16(
      false, a, false, b, (short)0, c, false, false);
}

// -------------------------------------------------------------------------
// h = x @ lin_in_w + b   ([N,5] @ [5,64])
// -------------------------------------------------------------------------
__global__ void k_lin_in(const float* __restrict__ x, const float* __restrict__ w,
                         const float* __restrict__ b, float* __restrict__ h) {
  int i = blockIdx.x * blockDim.x + threadIdx.x;   // i over N*64 (exact)
  int node = i >> 6, e = i & 63;
  const float* xr = x + node * 5;
  float acc = b[e];
#pragma unroll
  for (int k = 0; k < 5; ++k) acc += xr[k] * w[k * 64 + e];
  h[i] = acc;
}

// -------------------------------------------------------------------------
// Per-edge message MLP + scatter-add.  One wave = 16 edges.
// m = relu(relu([h_dst,h_src,dist] @ W1 + b1) @ W2 + b2); aggr[dst] += m
// -------------------------------------------------------------------------
__global__ void __launch_bounds__(128) k_msg(
    const float* __restrict__ h, const float* __restrict__ dists,
    const long long* __restrict__ esrc, const long long* __restrict__ edst,
    const float* __restrict__ w1, const float* __restrict__ b1,
    const float* __restrict__ w2, const float* __restrict__ b2,
    float* __restrict__ aggr) {
  __shared__ unsigned short W1t[64][136];   // col-major bf16, padded stride
  __shared__ unsigned short W2t[64][72];
  __shared__ float w128s[64];               // dist row of W1 (f32)
  __shared__ float b1s[64], b2s[64];
  __shared__ unsigned short mS[4][16][72];  // per-wave layer-1 output tile
  __shared__ float distS[4][16];
  __shared__ int   dstS[4][16];

  const int tid = threadIdx.x;
  for (int i = tid; i < 128 * 64; i += 128) {           // W1 K=0..127 transpose
    int k = i >> 6, n = i & 63;
    W1t[n][k] = bf1(w1[k * 64 + n]);
  }
  for (int i = tid; i < 64 * 64; i += 128) {
    int k = i >> 6, n = i & 63;
    W2t[n][k] = bf1(w2[k * 64 + n]);
  }
  if (tid < 64) {
    w128s[tid] = w1[128 * 64 + tid];
    b1s[tid] = b1[tid];
    b2s[tid] = b2[tid];
  }
  __syncthreads();

  const int w     = tid >> 5;
  const int lane  = tid & 31;
  const int row   = lane & 15;
  const int half  = lane >> 4;
  const int kbase = half * 8;
  const int rhalf = half * 8;
  const int tile  = blockIdx.x * 4 + w;     // NE/16 = 100000 tiles, exact grid

  const int e  = tile * 16 + row;
  const int dI = (int)edst[e];
  const int sI = (int)esrc[e];
  if (half == 0) {
    dstS[w][row]  = dI;
    distS[w][row] = dists[e];
  }

  // A fragments: 16 edges x 128 (h_dst || h_src), 4 chunks of K=32 in bf16
  v16bf afrag[4];
#pragma unroll
  for (int c = 0; c < 4; ++c) {
    const float* base = h + (c < 2 ? dI : sI) * 64 + (c & 1) * 32 + kbase;
    float4 f0 = *(const float4*)(base);
    float4 f1 = *(const float4*)(base + 4);
    float4 f2 = *(const float4*)(base + 16);
    float4 f3 = *(const float4*)(base + 20);
    afrag[c] = packA(f0, f1, f2, f3);
  }

  // layer 1: 4 N-tiles x 4 K-chunks of v_wmma_f32_16x16x32_bf16
  v8f acc[4] = {};
#pragma unroll
  for (int t = 0; t < 4; ++t) {
    const int n = t * 16 + row;             // this lane's B column
#pragma unroll
    for (int c = 0; c < 4; ++c)
      acc[t] = wmma_bf16(afrag[c], ldsB(&W1t[n][c * 32 + half * 8]), acc[t]);
  }

  // epilogue 1: + dist*w1[128,:] + b1, relu, stash tile as bf16 for layer 2 A
#pragma unroll
  for (int t = 0; t < 4; ++t) {
    const int col = t * 16 + row;
    const float wd = w128s[col], bb = b1s[col];
#pragma unroll
    for (int i = 0; i < 8; ++i) {
      float v = fmaxf(acc[t][i] + distS[w][rhalf + i] * wd + bb, 0.0f);
      mS[w][rhalf + i][col] = bf1(v);
    }
  }

  // layer 2 A fragments from LDS tile (same-wave LDS ops are in-order)
  v16bf a2[2];
#pragma unroll
  for (int c = 0; c < 2; ++c) {
    v8u16 lo = *(const v8u16*)&mS[w][row][c * 32 + kbase];
    v8u16 hi = *(const v8u16*)&mS[w][row][c * 32 + 16 + kbase];
    a2[c] = joinB(lo, hi);
  }

  v8f acc2[4] = {};
#pragma unroll
  for (int t = 0; t < 4; ++t) {
    const int n = t * 16 + row;
#pragma unroll
    for (int c = 0; c < 2; ++c)
      acc2[t] = wmma_bf16(a2[c], ldsB(&W2t[n][c * 32 + half * 8]), acc2[t]);
  }

  // epilogue 2: relu + L2-resident scatter-add into aggr[dst]
#pragma unroll
  for (int t = 0; t < 4; ++t) {
    const int col = t * 16 + row;
    const float bb = b2s[col];
#pragma unroll
    for (int i = 0; i < 8; ++i) {
      float v = fmaxf(acc2[t][i] + bb, 0.0f);
      atomicAdd(&aggr[dstS[w][rhalf + i] * 64 + col], v);
    }
  }
}

// -------------------------------------------------------------------------
// Per-node update MLP + residual.  One wave = 16 nodes.
// h += relu(relu([h,aggr] @ U1 + b1) @ U2 + b2)
// -------------------------------------------------------------------------
__global__ void __launch_bounds__(128) k_upd(
    float* __restrict__ h, const float* __restrict__ aggr,
    const float* __restrict__ w1, const float* __restrict__ b1,
    const float* __restrict__ w2, const float* __restrict__ b2, int ntiles) {
  __shared__ unsigned short W1t[64][136];
  __shared__ unsigned short W2t[64][72];
  __shared__ float b1s[64], b2s[64];
  __shared__ unsigned short uS[4][16][72];

  const int tid = threadIdx.x;
  for (int i = tid; i < 128 * 64; i += 128) {
    int k = i >> 6, n = i & 63;
    W1t[n][k] = bf1(w1[k * 64 + n]);
  }
  for (int i = tid; i < 64 * 64; i += 128) {
    int k = i >> 6, n = i & 63;
    W2t[n][k] = bf1(w2[k * 64 + n]);
  }
  if (tid < 64) { b1s[tid] = b1[tid]; b2s[tid] = b2[tid]; }
  __syncthreads();

  const int w = tid >> 5, lane = tid & 31;
  const int row = lane & 15, half = lane >> 4;
  const int kbase = half * 8, rhalf = half * 8;
  const int tile = blockIdx.x * 4 + w;
  if (tile >= ntiles) return;
  const int node = tile * 16 + row;

  v16bf afrag[4];
#pragma unroll
  for (int c = 0; c < 4; ++c) {
    const float* src  = (c < 2) ? h : aggr;
    const float* base = src + node * 64 + (c & 1) * 32 + kbase;
    float4 f0 = *(const float4*)(base);
    float4 f1 = *(const float4*)(base + 4);
    float4 f2 = *(const float4*)(base + 16);
    float4 f3 = *(const float4*)(base + 20);
    afrag[c] = packA(f0, f1, f2, f3);
  }

  v8f acc[4] = {};
#pragma unroll
  for (int t = 0; t < 4; ++t) {
    const int n = t * 16 + row;
#pragma unroll
    for (int c = 0; c < 4; ++c)
      acc[t] = wmma_bf16(afrag[c], ldsB(&W1t[n][c * 32 + half * 8]), acc[t]);
  }

#pragma unroll
  for (int t = 0; t < 4; ++t) {
    const int col = t * 16 + row;
    const float bb = b1s[col];
#pragma unroll
    for (int i = 0; i < 8; ++i)
      uS[w][rhalf + i][col] = bf1(fmaxf(acc[t][i] + bb, 0.0f));
  }

  v16bf a2[2];
#pragma unroll
  for (int c = 0; c < 2; ++c) {
    v8u16 lo = *(const v8u16*)&uS[w][row][c * 32 + kbase];
    v8u16 hi = *(const v8u16*)&uS[w][row][c * 32 + 16 + kbase];
    a2[c] = joinB(lo, hi);
  }

  v8f acc2[4] = {};
#pragma unroll
  for (int t = 0; t < 4; ++t) {
    const int n = t * 16 + row;
#pragma unroll
    for (int c = 0; c < 2; ++c)
      acc2[t] = wmma_bf16(a2[c], ldsB(&W2t[n][c * 32 + half * 8]), acc2[t]);
  }

  // residual: h += relu(u2 + b2)   (tiles disjoint -> plain read-modify-write)
#pragma unroll
  for (int t = 0; t < 4; ++t) {
    const int col = t * 16 + row;
    const float bb = b2s[col];
#pragma unroll
    for (int i = 0; i < 8; ++i) {
      int off = (tile * 16 + rhalf + i) * 64 + col;
      h[off] = h[off] + fmaxf(acc2[t][i] + bb, 0.0f);
    }
  }
}

// -------------------------------------------------------------------------
// Global mean pool partials (batch is all-zeros -> single graph)
// -------------------------------------------------------------------------
__global__ void k_pool(const float* __restrict__ h, float* __restrict__ pool,
                       int nnodes) {
  __shared__ float sp[64];
  int tid = threadIdx.x;
  if (tid < 64) sp[tid] = 0.0f;
  __syncthreads();
  int col = tid & 63, sub = tid >> 6;
  float s = 0.0f;
  for (int j = 0; j < 128; ++j) {
    int node = blockIdx.x * 512 + j * 4 + sub;
    if (node < nnodes) s += h[node * 64 + col];
  }
  atomicAdd(&sp[col], s);
  __syncthreads();
  if (tid < 64) atomicAdd(&pool[tid], sp[tid]);
}

__global__ void k_pred(const float* __restrict__ pool,
                       const float* __restrict__ pw,
                       const float* __restrict__ pb, float* __restrict__ out,
                       float inv_n) {
  float s = 0.0f;
#pragma unroll
  for (int c = 0; c < 64; ++c) s += pool[c] * pw[c];
  out[0] = s * inv_n + pb[0];
}

// -------------------------------------------------------------------------
extern "C" void kernel_launch(void* const* d_in, const int* in_sizes, int n_in,
                              void* d_out, int out_size, void* d_ws,
                              size_t ws_size, hipStream_t stream) {
  (void)in_sizes; (void)n_in; (void)out_size; (void)ws_size;
  const float*     x     = (const float*)d_in[0];
  const float*     dists = (const float*)d_in[1];
  const long long* eidx  = (const long long*)d_in[2];   // [2][E] int64
  // d_in[3] = batch (all zeros) -> single-graph mean pool
  const float* lin_w = (const float*)d_in[4];
  const float* lin_b = (const float*)d_in[5];
  const float* mw1   = (const float*)d_in[6];
  const float* mb1   = (const float*)d_in[7];
  const float* mw2   = (const float*)d_in[8];
  const float* mb2   = (const float*)d_in[9];
  const float* uw1   = (const float*)d_in[10];
  const float* ub1   = (const float*)d_in[11];
  const float* uw2   = (const float*)d_in[12];
  const float* ub2   = (const float*)d_in[13];
  const float* pw    = (const float*)d_in[14];
  const float* pb    = (const float*)d_in[15];

  float* h    = (float*)d_ws;          // [N,64]
  float* aggr = h + (size_t)NN * 64;   // [N,64]
  float* pool = aggr + (size_t)NN * 64;

  const long long* esrc = eidx;        // edge_index[0]
  const long long* edst = eidx + NE;   // edge_index[1]

  k_lin_in<<<(NN * 64) / 256, 256, 0, stream>>>(x, lin_w, lin_b, h);

  for (int l = 0; l < 4; ++l) {
    hipMemsetAsync(aggr, 0, (size_t)NN * 64 * sizeof(float), stream);
    k_msg<<<NE / 16 / 4, 128, 0, stream>>>(
        h, dists, esrc, edst, mw1 + l * 129 * 64, mb1 + l * 64,
        mw2 + l * 64 * 64, mb2 + l * 64, aggr);
    int ntiles = NN / 16;  // 6250
    k_upd<<<(ntiles + 3) / 4, 128, 0, stream>>>(
        h, aggr, uw1 + l * 128 * 64, ub1 + l * 64, uw2 + l * 64 * 64,
        ub2 + l * 64, ntiles);
  }

  hipMemsetAsync(pool, 0, 64 * sizeof(float), stream);
  k_pool<<<(NN + 511) / 512, 256, 0, stream>>>(h, pool, NN);
  k_pred<<<1, 1, 0, stream>>>(pool, pw, pb, (float*)d_out, 1.0f / NN);
}
